// MultimodalContrastiveModel_12180527252071
// MI455X (gfx1250) — matrix operand order repaired and checked by reading, not compiled, not run on hardware
//
#include <hip/hip_runtime.h>

#define DD      300
#define LDP     304
#define NNODES  50000
#define NEDGES  800000
#define NGRAPHS 256
#define NLAYERS 5
#define EMBD    600

#if defined(__has_builtin) && __has_builtin(__builtin_amdgcn_sched_barrier)
#define SCHED_FENCE() __builtin_amdgcn_sched_barrier(0)
#else
#define SCHED_FENCE()
#endif

typedef __attribute__((ext_vector_type(16))) _Float16 v16h;
typedef __attribute__((ext_vector_type(8)))  float    v8f;

union FragH { v16h v; _Float16 h[16]; };

// ---------------------------------------------------------------------------
// Templated WMMA GEMM:  Out[M,Ncols] = act( A[M,K=300] @ B[300,Ncols] + bias )
// MODE 0: A from global fp32 (lda)
// MODE 1: A = relu(edge_attr[row,0:3] @ W1f + b1f)   (fused tiny-K layer)
// MODE 2: A = relu([chir,fchg] @ W1f + b1f), epilogue += atom_emb[z[row]]
// Block: 256 thr = 8 waves (2 in M x 4 in N). Tile: 32 rows x NT*64 cols.
// All B fragments of a batch are DS-loaded before the WMMA chain (enforced
// with a sched_barrier) so the v_wmma ops issue back-to-back while later
// DS loads drain (non-zero dscnt waits instead of serial dscnt==0 stalls).
// ---------------------------------------------------------------------------
template<int NT, int MODE>
__global__ __launch_bounds__(256)
void k_wmma_gemm(int M, int Ncols,
                 const float* __restrict__ A, int lda,
                 const float* __restrict__ B, int ldb,
                 const float* __restrict__ bias,
                 float* __restrict__ Out, int ldo,
                 int relu_out,
                 const float* __restrict__ attr0,
                 const float* __restrict__ attr1,
                 const float* __restrict__ W1f,
                 const float* __restrict__ b1f,
                 const int*   __restrict__ zidx,
                 const float* __restrict__ emb)
{
    constexpr int NSPAN = NT * 64;            // cols covered per block
    constexpr int TB    = (NT < 5) ? NT : 5;  // B-frag batch size
    __shared__ _Float16 As[32 * 34];          // [row][k], stride 34 (bank-spread)
    __shared__ _Float16 Bs[NSPAN * 34];       // [col][k], stride 34

    const int tid  = threadIdx.x;
    const int wave = tid >> 5;
    const int lane = tid & 31;
    const int wm   = wave & 1;                // M-wave (0..1)
    const int wn   = wave >> 1;               // N-wave (0..3)
    const int g    = lane >> 4;               // lane half
    const int ln16 = lane & 15;
    const int row0 = blockIdx.x * 32;

    v8f vzero = {};
    v8f acc[NT];
    for (int t = 0; t < NT; ++t) acc[t] = vzero;

    for (int k0 = 0; k0 < DD; k0 += 32) {
        __syncthreads();                      // previous iteration's reads done

        // prefetch next K-chunk toward L2 / WGP$ (global_prefetch_b8)
        if (k0 + 32 < DD) {
            if (MODE == 0) {
                int pr = row0 + (tid >> 3);
                if (pr < M) __builtin_prefetch(&A[(size_t)pr * lda + k0 + 32 + (tid & 7) * 4], 0, 1);
            }
            int pk = k0 + 32 + (tid >> 5);
            __builtin_prefetch(&B[(size_t)pk * ldb + (tid & 31) * 8], 0, 1);
        }

        // ---- stage A tile: 32 rows x 32 k (fp32 -> fp16) ----
        for (int i = tid; i < 32 * 32; i += 256) {
            int r  = i >> 5, kk = i & 31;
            int row = row0 + r;
            int k   = k0 + kk;
            float v = 0.f;
            if (row < M && k < DD) {
                if (MODE == 0) {
                    v = A[(size_t)row * lda + k];
                } else if (MODE == 1) {
                    const float* ea = attr0 + (size_t)row * 3;
                    float h = b1f[k] + ea[0] * W1f[k] + ea[1] * W1f[DD + k]
                                     + ea[2] * W1f[2 * DD + k];
                    v = fmaxf(h, 0.f);
                } else {
                    float h = b1f[k] + attr0[row] * W1f[k] + attr1[row] * W1f[DD + k];
                    v = fmaxf(h, 0.f);
                }
            }
            As[r * 34 + kk] = (_Float16)v;
        }
        // ---- stage B tile: NSPAN cols x 32 k, stored [col][k] ----
        for (int kk = 0; kk < 32; ++kk) {
            int k = k0 + kk;
            const float* Brow = B + (size_t)k * ldb;
            for (int col = tid; col < NSPAN; col += 256) {
                float v = (k < DD && col < Ncols) ? Brow[col] : 0.f;
                Bs[col * 34 + kk] = (_Float16)v;
            }
        }
        __syncthreads();

        // ---- build A fragment (16x32, ISA 7.12.2 layout) ----
        FragH a;
        {
            const _Float16* base = &As[(wm * 16 + ln16) * 34];
            #pragma unroll
            for (int v2 = 0; v2 < 8; ++v2) {
                int kb = ((v2 < 4) ? 0 : 16) + g * 8 + (v2 & 3) * 2;
                a.h[2 * v2]     = base[kb];
                a.h[2 * v2 + 1] = base[kb + 1];
            }
        }
        // ---- N tiles in batches: load TB B-frags, fence, TB b2b WMMAs ----
        #pragma unroll
        for (int tb = 0; tb < NT; tb += TB) {
            FragH bf[TB];
            #pragma unroll
            for (int t = 0; t < TB; ++t) {
                int ct = (wn * NT + tb + t) * 16;
                const _Float16* cb = &Bs[(ct + ln16) * 34];
                #pragma unroll
                for (int v2 = 0; v2 < 8; ++v2) {
                    int kb = ((v2 < 4) ? 0 : 16) + g * 8 + (v2 & 3) * 2;
                    bf[t].h[2 * v2]     = cb[kb];
                    bf[t].h[2 * v2 + 1] = cb[kb + 1];
                }
            }
            SCHED_FENCE();   // all DS loads issued before any WMMA below
            #pragma unroll
            for (int t = 0; t < TB; ++t) {
                acc[tb + t] = __builtin_amdgcn_wmma_f32_16x16x32_f16(
                    false, a.v, false, bf[t].v, (short)0, acc[tb + t], false, false);
            }
            SCHED_FENCE();
        }
    }

    // ---- epilogue: bias (+emb gather) (+relu), guarded stores ----
    for (int t = 0; t < NT; ++t) {
        int col = (wn * NT + t) * 16 + ln16;
        if (col >= Ncols) continue;
        float bv = bias ? bias[col] : 0.f;
        #pragma unroll
        for (int v2 = 0; v2 < 8; ++v2) {
            int row = row0 + wm * 16 + g * 8 + v2;
            if (row >= M) continue;
            float val = acc[t][v2] + bv;
            if (MODE == 2) val += emb[(size_t)zidx[row] * DD + col];
            if (relu_out)  val = fmaxf(val, 0.f);
            Out[(size_t)row * ldo + col] = val;
        }
    }
}

// ---------------------------------------------------------------------------
// msg = relu(x[src] + e); agg[dst] += msg   (agg pre-initialized to x)
// thread per (edge, 16B column chunk): 75 chunks of float4 per edge
// ---------------------------------------------------------------------------
__global__ __launch_bounds__(256)
void k_msg_agg(const int* __restrict__ ei,
               const float* __restrict__ x,
               const float* __restrict__ e,
               float* __restrict__ agg)
{
    long idx = (long)blockIdx.x * 256 + threadIdx.x;   // NEDGES*75 total, exact
    int eid = (int)(idx / 75);
    int c4  = (int)(idx % 75);
    int src = ei[eid];
    int dst = ei[NEDGES + eid];
    const float4 xv = *(const float4*)&x[(size_t)src * LDP + c4 * 4];
    const float4 ev = *(const float4*)&e[(size_t)eid * LDP + c4 * 4];
    float4 m;
    m.x = fmaxf(xv.x + ev.x, 0.f);
    m.y = fmaxf(xv.y + ev.y, 0.f);
    m.z = fmaxf(xv.z + ev.z, 0.f);
    m.w = fmaxf(xv.w + ev.w, 0.f);
    float* ap = &agg[(size_t)dst * LDP + c4 * 4];
    atomicAdd(ap + 0, m.x);
    atomicAdd(ap + 1, m.y);
    atomicAdd(ap + 2, m.z);
    atomicAdd(ap + 3, m.w);
}

// ---------------------------------------------------------------------------
// x = LayerNorm(x + relu(h2)) * g + b, one wave32 per row
// ---------------------------------------------------------------------------
__global__ __launch_bounds__(256)
void k_ln(const float* __restrict__ h2, float* __restrict__ x,
          const float* __restrict__ gam, const float* __restrict__ bet)
{
    int row  = blockIdx.x * 8 + (threadIdx.x >> 5);
    int lane = threadIdx.x & 31;
    if (row >= NNODES) return;
    const float* xr = x  + (size_t)row * LDP;
    const float* hr = h2 + (size_t)row * LDP;
    float t[10];
    float s1 = 0.f, s2 = 0.f;
    int cnt = 0;
    for (int c = lane; c < DD; c += 32) {
        float v = xr[c] + fmaxf(hr[c], 0.f);
        t[cnt++] = v;
        s1 += v;
        s2 += v * v;
    }
    for (int o = 16; o > 0; o >>= 1) {
        s1 += __shfl_xor(s1, o, 32);
        s2 += __shfl_xor(s2, o, 32);
    }
    float mu  = s1 * (1.f / 300.f);
    float var = s2 * (1.f / 300.f) - mu * mu;
    float r   = rsqrtf(var + 1e-5f);
    cnt = 0;
    for (int c = lane; c < DD; c += 32) {
        float v = t[cnt++];
        x[(size_t)row * LDP + c] = (v - mu) * r * gam[c] + bet[c];
    }
}

// ---------------------------------------------------------------------------
// segment-sum pooling: sums[batch[n]] += x[n,:]; counts[batch[n]] += 1
// ---------------------------------------------------------------------------
__global__ __launch_bounds__(256)
void k_pool_sum(const int* __restrict__ batch, const float* __restrict__ x,
                float* __restrict__ ps, float* __restrict__ cnt)
{
    long idx = (long)blockIdx.x * 256 + threadIdx.x;
    if (idx >= (long)NNODES * 75) return;
    int node = (int)(idx / 75);
    int c4   = (int)(idx % 75);
    int gph  = batch[node];
    const float4 v = *(const float4*)&x[(size_t)node * LDP + c4 * 4];
    float* sp = &ps[(size_t)gph * LDP + c4 * 4];
    atomicAdd(sp + 0, v.x);
    atomicAdd(sp + 1, v.y);
    atomicAdd(sp + 2, v.z);
    atomicAdd(sp + 3, v.w);
    if (c4 == 0) atomicAdd(&cnt[gph], 1.f);
}

__global__ __launch_bounds__(256)
void k_pool_div(const float* __restrict__ ps, const float* __restrict__ cnt,
                float* __restrict__ p1)
{
    int idx = blockIdx.x * 256 + threadIdx.x;
    if (idx >= NGRAPHS * DD) return;
    int gph = idx / DD, c = idx % DD;
    p1[(size_t)gph * LDP + c] = ps[(size_t)gph * LDP + c] / fmaxf(cnt[gph], 1.f);
}

// ---------------------------------------------------------------------------
extern "C" void kernel_launch(void* const* d_in, const int* in_sizes, int n_in,
                              void* d_out, int out_size, void* d_ws, size_t ws_size,
                              hipStream_t stream)
{
    const int*   z        = (const int*)  d_in[0];
    const float* chir     = (const float*)d_in[1];
    const float* fchg     = (const float*)d_in[2];
    const int*   eidx     = (const int*)  d_in[3];
    const float* eattr    = (const float*)d_in[4];
    const int*   batch    = (const int*)  d_in[5];
    const float* atom_emb = (const float*)d_in[6];
    const float* nap_W1   = (const float*)d_in[7];
    const float* nap_b1   = (const float*)d_in[8];
    const float* nap_W2   = (const float*)d_in[9];
    const float* nap_b2   = (const float*)d_in[10];
    const float* ee_W1    = (const float*)d_in[11];
    const float* ee_b1    = (const float*)d_in[12];
    const float* ee_W2    = (const float*)d_in[13];
    const float* ee_b2    = (const float*)d_in[14];
    const float* gine_W1  = (const float*)d_in[15];
    const float* gine_b1  = (const float*)d_in[16];
    const float* gine_W2  = (const float*)d_in[17];
    const float* gine_b2  = (const float*)d_in[18];
    const float* ln_g     = (const float*)d_in[19];
    const float* ln_b     = (const float*)d_in[20];
    const float* pool_W   = (const float*)d_in[21];
    const float* pool_b   = (const float*)d_in[22];
    const float* proj_W   = (const float*)d_in[23];
    const float* proj_b   = (const float*)d_in[24];
    float* out = (float*)d_out;

    float* ws = (float*)d_ws;
    size_t off = 0;
    float* e_buf = ws + off; off += (size_t)NEDGES  * LDP;
    float* x     = ws + off; off += (size_t)NNODES  * LDP;
    float* agg   = ws + off; off += (size_t)NNODES  * LDP;
    float* h1    = ws + off; off += (size_t)NNODES  * LDP;
    float* h2    = ws + off; off += (size_t)NNODES  * LDP;
    float* ps    = ws + off; off += (size_t)NGRAPHS * LDP;
    float* cnt   = ws + off; off += (size_t)NGRAPHS;
    float* p1    = ws + off; off += (size_t)NGRAPHS * LDP;
    float* p2    = ws + off;

    dim3 blk(256);

    // edge embedding: e = relu(edge_attr@ee_W1+b1)@ee_W2+b2   (first layer fused)
    k_wmma_gemm<5, 1><<<(NEDGES + 31) / 32, blk, 0, stream>>>(
        NEDGES, DD, nullptr, 0, ee_W2, DD, ee_b2, e_buf, LDP, 0,
        eattr, nullptr, ee_W1, ee_b1, nullptr, nullptr);

    // node features: x = atom_emb[z] + relu(attr@nap_W1+b1)@nap_W2+b2
    k_wmma_gemm<5, 2><<<(NNODES + 31) / 32, blk, 0, stream>>>(
        NNODES, DD, nullptr, 0, nap_W2, DD, nap_b2, x, LDP, 0,
        chir, fchg, nap_W1, nap_b1, z, atom_emb);

    for (int l = 0; l < NLAYERS; ++l) {
        // agg = x  (so agg + sum(msg) comes out of the scatter directly)
        hipMemcpyAsync(agg, x, (size_t)NNODES * LDP * sizeof(float),
                       hipMemcpyDeviceToDevice, stream);
        k_msg_agg<<<(int)(((long)NEDGES * 75) / 256), blk, 0, stream>>>(
            eidx, x, e_buf, agg);
        // h1 = relu(agg @ W1 + b1)
        k_wmma_gemm<5, 0><<<(NNODES + 31) / 32, blk, 0, stream>>>(
            NNODES, DD, agg, LDP, gine_W1 + (size_t)l * DD * DD, DD,
            gine_b1 + (size_t)l * DD, h1, LDP, 1,
            nullptr, nullptr, nullptr, nullptr, nullptr, nullptr);
        // h2 = h1 @ W2 + b2
        k_wmma_gemm<5, 0><<<(NNODES + 31) / 32, blk, 0, stream>>>(
            NNODES, DD, h1, LDP, gine_W2 + (size_t)l * DD * DD, DD,
            gine_b2 + (size_t)l * DD, h2, LDP, 0,
            nullptr, nullptr, nullptr, nullptr, nullptr, nullptr);
        // x = LN(x + relu(h2))
        k_ln<<<(NNODES + 7) / 8, blk, 0, stream>>>(
            h2, x, ln_g + (size_t)l * DD, ln_b + (size_t)l * DD);
    }

    // mean pooling per graph
    hipMemsetAsync(ps, 0, ((size_t)NGRAPHS * LDP + NGRAPHS) * sizeof(float), stream);
    k_pool_sum<<<(int)(((long)NNODES * 75 + 255) / 256), blk, 0, stream>>>(
        batch, x, ps, cnt);
    k_pool_div<<<(NGRAPHS * DD + 255) / 256, blk, 0, stream>>>(ps, cnt, p1);

    // pooled @ pool_W + pool_b
    k_wmma_gemm<5, 0><<<(NGRAPHS + 31) / 32, blk, 0, stream>>>(
        NGRAPHS, DD, p1, LDP, pool_W, DD, pool_b, p2, LDP, 0,
        nullptr, nullptr, nullptr, nullptr, nullptr, nullptr);
    // @ proj_W + proj_b -> d_out [256,600]
    k_wmma_gemm<10, 0><<<(NGRAPHS + 31) / 32, blk, 0, stream>>>(
        NGRAPHS, EMBD, p2, LDP, proj_W, EMBD, proj_b, out, EMBD, 0,
        nullptr, nullptr, nullptr, nullptr, nullptr, nullptr);
}
